// Attention_68470368633621
// MI455X (gfx1250) — matrix operand order repaired and checked by reading, not compiled, not run on hardware
//
#include <hip/hip_runtime.h>
#include <hip/hip_bf16.h>
#include <stdint.h>

// Problem constants (match reference)
#define B_  2
#define SQ_ 2048
#define SK_ 2048
#define H_  1024
#define NH_ 16
#define D_  64
#define KC_ 3072   // conv-as-GEMM K = 3 * 1024

typedef __attribute__((ext_vector_type(16))) __bf16 v16bf;
typedef __attribute__((ext_vector_type(8)))  __bf16 v8bf;
typedef __attribute__((ext_vector_type(8)))  float  v8f;
typedef __attribute__((ext_vector_type(4)))  uint32_t v4u;
typedef __attribute__((ext_vector_type(8)))  int      v8i;
typedef __attribute__((ext_vector_type(4)))  int      v4i;

// ---------- helpers ----------

__device__ __forceinline__ __bf16 f2bf(float f) {
  union { float f; uint32_t u; } x; x.f = f;
  uint32_t r = (x.u + 0x7FFFu + ((x.u >> 16) & 1u)) >> 16;  // RNE
  union { uint16_t u; __bf16 b; } y; y.u = (uint16_t)r;
  return y.b;
}

__device__ __forceinline__ v8bf  ld8 (const __bf16* p) { return *reinterpret_cast<const v8bf*>(p); }
__device__ __forceinline__ v16bf ld16(const __bf16* p) { return *reinterpret_cast<const v16bf*>(p); }

__device__ __forceinline__ v16bf cat8(v8bf lo, v8bf hi) {
  v16bf r;
#pragma unroll
  for (int i = 0; i < 8; ++i) { r[i] = lo[i]; r[i + 8] = hi[i]; }
  return r;
}

__device__ __forceinline__ v16bf zero16() {
  v16bf r;
  union { uint16_t u; __bf16 b; } z; z.u = 0;
#pragma unroll
  for (int i = 0; i < 16; ++i) r[i] = z.b;
  return r;
}

__device__ __forceinline__ v8f wmma_bf16(v16bf a, v16bf b, v8f c) {
  // D = A(16x32 bf16) * B(32x16 bf16) + C(16x16 f32)
  return __builtin_amdgcn_wmma_f32_16x16x32_bf16(
      /*neg_a=*/false, a, /*neg_b=*/false, b,
      /*c_mod=*/(short)0, c, /*reuse_a=*/false, /*reuse_b=*/false);
}

// reductions across the 16-lane group holding one C-matrix row
__device__ __forceinline__ float rmax16(float v) {
#pragma unroll
  for (int m = 1; m < 16; m <<= 1) v = fmaxf(v, __shfl_xor(v, m, 32));
  return v;
}
__device__ __forceinline__ float rsum16(float v) {
#pragma unroll
  for (int m = 1; m < 16; m <<= 1) v += __shfl_xor(v, m, 32);
  return v;
}

// ---------- Tensor Data Mover: 2D tile (2-byte elements) global -> LDS ----------
// D# per CDNA5 ISA ch.8: group0 {count=1, lds_addr, global_addr(57b), type=2},
// group1 {data_size=1, tensor_dim0/1, tile_dim0/1, tensor_dim0_stride}.
__device__ __forceinline__ void tdm_load_2d_bf16(uint32_t lds_off_bytes, const void* gptr,
                                                 uint32_t tensor_d0, uint32_t tensor_d1,
                                                 uint32_t tile_d0, uint32_t tile_d1,
                                                 uint64_t stride0) {
  uint64_t ga = (uint64_t)(uintptr_t)gptr;
  v4u g0;
  g0[0] = 1u;                                   // count=1, user mode, no gather
  g0[1] = lds_off_bytes;                        // lds_addr (bytes)
  g0[2] = (uint32_t)ga;                         // global_addr[31:0]
  g0[3] = (uint32_t)(ga >> 32) | (2u << 30);    // global_addr[56:32] | type=2
  v8i g1;
  g1[0] = (int)(1u << 16);                      // workgroup_mask=0, data_size=1 (2B)
  g1[1] = (int)((tensor_d0 & 0xFFFFu) << 16);   // tensor_dim0[15:0] @ bits 63:48
  g1[2] = (int)((tensor_d0 >> 16) | ((tensor_d1 & 0xFFFFu) << 16));
  g1[3] = (int)((tensor_d1 >> 16) | (tile_d0 << 16));
  g1[4] = (int)(tile_d1 & 0xFFFFu);             // tile_dim1; tile_dim2=0 (2D)
  g1[5] = (int)(uint32_t)(stride0 & 0xFFFFFFFFu);
  g1[6] = (int)(uint32_t)((stride0 >> 32) & 0xFFFFu);  // dim1_stride unused (2D)
  g1[7] = 0;
  v4i z4 = {0, 0, 0, 0};
#if defined(__clang_major__) && (__clang_major__ >= 23)
  v8i z8 = {};
  __builtin_amdgcn_tensor_load_to_lds(g0, g1, z4, z4, z8, 0);
#else
  __builtin_amdgcn_tensor_load_to_lds(g0, g1, z4, z4, 0);
#endif
}

// ---------- stage 0: conversions ----------

__global__ __launch_bounds__(256) void k_cvt_bf16(const float* __restrict__ s,
                                                  __bf16* __restrict__ d, int n) {
  int i = blockIdx.x * 256 + threadIdx.x;
  if (i < n) d[i] = f2bf(s[i]);
}

// W: [taps][H][H] (tap, in, out) f32  ->  Wt: [H_out][taps*H] bf16 (transposed)
__global__ __launch_bounds__(256) void k_transpose_w(const float* __restrict__ W,
                                                     __bf16* __restrict__ Wt, int KH) {
  int tid = blockIdx.x * 256 + threadIdx.x;                 // tid = n*KH + kk
  int n = tid / KH;
  int kk = tid - n * KH;                                    // kk = tap*H + i
  Wt[tid] = f2bf(W[(size_t)kk * H_ + n]);
}

// ---------- stage 1: QKV conv-projection as WMMA GEMM (software-pipelined) ----------

__device__ __forceinline__ v16bf load_a_conv(const __bf16* Xb, int m0, int r, int h, int kk0) {
  const int tap = kk0 >> 10;             // K chunk of 32 never crosses a conv tap
  const int i0  = kk0 & 1023;
  const int t   = m0 + r + tap - 1;      // 'same' padding offset
  if ((unsigned)t < (unsigned)SQ_) {
    const __bf16* px = Xb + (size_t)t * H_ + i0 + h * 16;  // chunks at h*16 and h*16+8
    return cat8(ld8(px), ld8(px + 8));
  }
  return zero16();
}

// vmode 0: Out[b][head][t][d] (Q scaled / K), vmode 1: Out[b][head][d][t] (V transposed)
__global__ __launch_bounds__(256) void k_qkv_gemm(const __bf16* __restrict__ X,
                                                  const __bf16* __restrict__ Wt,
                                                  const float* __restrict__ bias,
                                                  __bf16* __restrict__ Out,
                                                  float scale, int vmode) {
  const int wave = threadIdx.x >> 5;
  const int lane = threadIdx.x & 31;
  const int h = lane >> 4;
  const int r = lane & 15;
  const int m0 = blockIdx.x * 128 + wave * 16;
  const int n0 = blockIdx.y * 64;
  const int b  = blockIdx.z;
  const __bf16* Xb = X + (size_t)b * SQ_ * H_;

  v8f acc[4] = {{}, {}, {}, {}};

  // prologue: fragments for kk0 = 0
  v16bf a = load_a_conv(Xb, m0, r, h, 0);
  v16bf bfr[4];
#pragma unroll
  for (int nc = 0; nc < 4; ++nc)
    bfr[nc] = ld16(Wt + (size_t)(n0 + nc * 16 + r) * KC_ + h * 16);

  for (int kk0 = 0; kk0 < KC_; kk0 += 32) {
    // prefetch next iteration's fragments before issuing this iteration's WMMAs
    v16bf an = a, bn[4] = {bfr[0], bfr[1], bfr[2], bfr[3]};
    if (kk0 + 32 < KC_) {
      an = load_a_conv(Xb, m0, r, h, kk0 + 32);
#pragma unroll
      for (int nc = 0; nc < 4; ++nc)
        bn[nc] = ld16(Wt + (size_t)(n0 + nc * 16 + r) * KC_ + (kk0 + 32) + h * 16);
    }
#pragma unroll
    for (int nc = 0; nc < 4; ++nc)
      acc[nc] = wmma_bf16(a, bfr[nc], acc[nc]);
    a = an;
#pragma unroll
    for (int nc = 0; nc < 4; ++nc) bfr[nc] = bn[nc];
  }

#pragma unroll
  for (int nc = 0; nc < 4; ++nc) {
    const int n = n0 + nc * 16 + r;
    const float bv = bias[n];
    const int head = n >> 6, d = n & 63;
#pragma unroll
    for (int j = 0; j < 8; ++j) {          // C-layout: VGPR j -> row 8h+j, col = lane&15
      const int t = m0 + 8 * h + j;
      float v = (acc[nc][j] + bv) * scale;
      size_t off = vmode
          ? ((((size_t)b * NH_ + head) * D_ + d) * SK_ + t)     // V transposed
          : ((((size_t)b * NH_ + head) * SQ_ + t) * D_ + d);    // Q / K
      Out[off] = f2bf(v);
    }
  }
}

// ---------- stage 2: flash attention with TDM double-buffered K/V tiles ----------
// Dynamic LDS layout (base offset 0):
//   [0      .. 16384)  K tiles  [2][64][64] bf16   (double buffer)
//   [16384  .. 32768)  V tiles  [2][64][64] bf16   (rows = d, cols = key)
//   [32768  .. 40960)  P tiles  [4][16][64] bf16   (per-wave C->A layout bridge)
__global__ __launch_bounds__(128) void k_attn(const __bf16* __restrict__ Qb,   // [B][NH][SQ][D]
                                              const __bf16* __restrict__ Kb,   // [B][NH][SK][D]
                                              const __bf16* __restrict__ Vt,   // [B][NH][D][SK]
                                              const float* __restrict__ mask,  // [B][SK]
                                              __bf16* __restrict__ AO) {       // [B][SQ][H]
  extern __shared__ __align__(32) char smem[];

  const int wave = threadIdx.x >> 5;
  const int lane = threadIdx.x & 31;
  const int h = lane >> 4;
  const int r = lane & 15;
  const int q0 = blockIdx.x * 64 + wave * 16;
  const int head = blockIdx.y, b = blockIdx.z;

  const __bf16* Qp = Qb + (((size_t)b * NH_ + head) * SQ_) * D_;
  const __bf16* Kp = Kb + (((size_t)b * NH_ + head) * SK_) * D_;
  const __bf16* Vp = Vt + (((size_t)b * NH_ + head) * D_) * SK_;
  const float*  mp = mask + (size_t)b * SK_;
  __bf16* P = (__bf16*)(smem + 32768 + wave * 2048);

  // Q A-fragments (Q pre-scaled by D^-0.5)
  v16bf qa[2];
#pragma unroll
  for (int kq = 0; kq < 2; ++kq) {
    const __bf16* p = Qp + (size_t)(q0 + r) * D_ + kq * 32 + h * 16;
    qa[kq] = cat8(ld8(p), ld8(p + 8));
  }

  float mi[8], li[8];
  v8f o[4] = {{}, {}, {}, {}};
#pragma unroll
  for (int j = 0; j < 8; ++j) { mi[j] = -3.0e38f; li[j] = 0.f; }

  const float L2E = 1.4426950408889634f;
  const int NT = SK_ / 64;

  // prologue: TDM-fetch tile 0 into buffer 0 (one wave drives the DMA engine)
  if (wave == 0) {
    tdm_load_2d_bf16(0u,     Kp, D_, SK_, 64, 64, D_);          // K[key0..][0..63]
    tdm_load_2d_bf16(16384u, Vp, SK_, D_, 64, 64, SK_);         // Vt[0..63][key0..]
    __builtin_amdgcn_s_wait_tensorcnt((short)0);
  }
  __syncthreads();

  for (int tile = 0; tile < NT; ++tile) {
    const int key0 = tile * 64;
    const int buf  = tile & 1;

    // async prefetch of next tile into the other buffer while we compute
    if (wave == 0 && tile + 1 < NT) {
      const int kn = key0 + 64;
      tdm_load_2d_bf16((uint32_t)((buf ^ 1) * 8192),
                       Kp + (size_t)kn * D_, D_, SK_, 64, 64, D_);
      tdm_load_2d_bf16((uint32_t)(16384 + (buf ^ 1) * 8192),
                       Vp + kn, SK_, D_, 64, 64, SK_);
    }
    const __bf16* kt = (const __bf16*)(smem + buf * 8192);            // [key][d]
    const __bf16* vt = (const __bf16*)(smem + 16384 + buf * 8192);    // [d][key]

    // ---- S = Q K^T for 16 rows x 64 keys (K B-frags from LDS) ----
    v8f s[4] = {{}, {}, {}, {}};
#pragma unroll
    for (int kq = 0; kq < 2; ++kq)
#pragma unroll
      for (int nc = 0; nc < 4; ++nc) {
        const __bf16* pk = kt + (nc * 16 + r) * D_ + kq * 32 + h * 16;
        s[nc] = wmma_bf16(qa[kq], ld16(pk), s[nc]);
      }

    // ---- additive mask ----
#pragma unroll
    for (int nc = 0; nc < 4; ++nc) {
      float mv = mp[key0 + nc * 16 + r] * -1e9f;
#pragma unroll
      for (int j = 0; j < 8; ++j) s[nc][j] += mv;
    }

    // ---- online softmax (rows 8h+j live in lanes [16h,16h+15]) ----
    float mnew[8], corr[8], rs[8];
#pragma unroll
    for (int j = 0; j < 8; ++j) {
      float tm = s[0][j];
#pragma unroll
      for (int nc = 1; nc < 4; ++nc) tm = fmaxf(tm, s[nc][j]);
      tm = rmax16(tm);
      mnew[j] = fmaxf(mi[j], tm);
      corr[j] = exp2f((mi[j] - mnew[j]) * L2E);
      rs[j] = 0.f;
    }
#pragma unroll
    for (int nc = 0; nc < 4; ++nc)
#pragma unroll
      for (int j = 0; j < 8; ++j) {
        float p_ = exp2f((s[nc][j] - mnew[j]) * L2E);
        s[nc][j] = p_;
        rs[j] += p_;
      }
#pragma unroll
    for (int j = 0; j < 8; ++j) {
      rs[j] = rsum16(rs[j]);
      li[j] = li[j] * corr[j] + rs[j];
      mi[j] = mnew[j];
    }
#pragma unroll
    for (int dc = 0; dc < 4; ++dc)
#pragma unroll
      for (int j = 0; j < 8; ++j) o[dc][j] *= corr[j];

    // ---- P (C layout) -> LDS -> A layout ----
#pragma unroll
    for (int nc = 0; nc < 4; ++nc)
#pragma unroll
      for (int j = 0; j < 8; ++j)
        P[(8 * h + j) * 64 + nc * 16 + r] = f2bf(s[nc][j]);

    // ---- O += P V (V B-frags from LDS, contiguous in key dim) ----
#pragma unroll
    for (int k0 = 0; k0 < 64; k0 += 32) {
      const __bf16* pp = P + r * 64 + k0 + h * 16;
      v16bf pa = cat8(ld8(pp), ld8(pp + 8));
#pragma unroll
      for (int dc = 0; dc < 4; ++dc) {
        const __bf16* pv = vt + (dc * 16 + r) * 64 + k0 + h * 16;
        o[dc] = wmma_bf16(pa, ld16(pv), o[dc]);
      }
    }

    // make next buffer ready & visible before anyone reads it
    if (wave == 0 && tile + 1 < NT) __builtin_amdgcn_s_wait_tensorcnt((short)0);
    __syncthreads();
  }

  // ---- normalize + merge heads: AO[b][row][head*64 + d] ----
#pragma unroll
  for (int dc = 0; dc < 4; ++dc)
#pragma unroll
    for (int j = 0; j < 8; ++j) {
      int row = q0 + 8 * h + j;
      int col = head * 64 + dc * 16 + r;
      AO[((size_t)b * SQ_ + row) * H_ + col] = f2bf(o[dc][j] / li[j]);
    }
}

// ---------- stage 3: output projection (M=4096, N=K=1024), pipelined ----------

__global__ __launch_bounds__(256) void k_out_gemm(const __bf16* __restrict__ Ain,  // [4096][1024]
                                                  const __bf16* __restrict__ Wt,   // [1024][1024] W^T
                                                  const float* __restrict__ bo,
                                                  float* __restrict__ out) {       // [4096][1024]
  const int wave = threadIdx.x >> 5;
  const int lane = threadIdx.x & 31;
  const int h = lane >> 4;
  const int r = lane & 15;
  const int m0 = blockIdx.x * 128 + wave * 16;
  const int n0 = blockIdx.y * 64;

  v8f acc[4] = {{}, {}, {}, {}};
  const __bf16* pa0 = Ain + (size_t)(m0 + r) * H_ + h * 16;
  v16bf a = cat8(ld8(pa0), ld8(pa0 + 8));
  v16bf bfr[4];
#pragma unroll
  for (int nc = 0; nc < 4; ++nc)
    bfr[nc] = ld16(Wt + (size_t)(n0 + nc * 16 + r) * H_ + h * 16);

  for (int kk0 = 0; kk0 < H_; kk0 += 32) {
    v16bf an = a, bn[4] = {bfr[0], bfr[1], bfr[2], bfr[3]};
    if (kk0 + 32 < H_) {
      const __bf16* pa = Ain + (size_t)(m0 + r) * H_ + (kk0 + 32) + h * 16;
      an = cat8(ld8(pa), ld8(pa + 8));
#pragma unroll
      for (int nc = 0; nc < 4; ++nc)
        bn[nc] = ld16(Wt + (size_t)(n0 + nc * 16 + r) * H_ + (kk0 + 32) + h * 16);
    }
#pragma unroll
    for (int nc = 0; nc < 4; ++nc)
      acc[nc] = wmma_bf16(a, bfr[nc], acc[nc]);
    a = an;
#pragma unroll
    for (int nc = 0; nc < 4; ++nc) bfr[nc] = bn[nc];
  }
#pragma unroll
  for (int nc = 0; nc < 4; ++nc) {
    const int n = n0 + nc * 16 + r;
    const float bv = bo[n];
#pragma unroll
    for (int j = 0; j < 8; ++j)
      out[(size_t)(m0 + 8 * h + j) * H_ + n] = acc[nc][j] + bv;
  }
}

// ---------- launch ----------

extern "C" void kernel_launch(void* const* d_in, const int* in_sizes, int n_in,
                              void* d_out, int out_size, void* d_ws, size_t ws_size,
                              hipStream_t stream) {
  const float* xq   = (const float*)d_in[0];
  const float* xs   = (const float*)d_in[1];
  const float* mask = (const float*)d_in[2];
  const float* Wq   = (const float*)d_in[3];
  const float* bq   = (const float*)d_in[4];
  const float* Wk   = (const float*)d_in[5];
  const float* bk   = (const float*)d_in[6];
  const float* Wv   = (const float*)d_in[7];
  const float* bv   = (const float*)d_in[8];
  const float* Wo   = (const float*)d_in[9];
  const float* bo   = (const float*)d_in[10];
  float* out = (float*)d_out;

  // workspace layout (bytes); total 68 MiB
  char* ws = (char*)d_ws;
  const size_t MB = (size_t)1 << 20;
  __bf16* xqb = (__bf16*)(ws + 0 * MB);    // [B][SQ][H]        8 MiB
  __bf16* xsb = (__bf16*)(ws + 8 * MB);    // [B][SK][H]        8 MiB
  __bf16* wqt = (__bf16*)(ws + 16 * MB);   // [H][3H]           6 MiB
  __bf16* wkt = (__bf16*)(ws + 22 * MB);   // [H][3H]           6 MiB
  __bf16* wvt = (__bf16*)(ws + 28 * MB);   // [H][3H]           6 MiB
  __bf16* wot = (__bf16*)(ws + 34 * MB);   // [H][H]            2 MiB
  __bf16* qb  = (__bf16*)(ws + 36 * MB);   // [B][NH][SQ][D]    8 MiB
  __bf16* kb  = (__bf16*)(ws + 44 * MB);   // [B][NH][SK][D]    8 MiB
  __bf16* vtb = (__bf16*)(ws + 52 * MB);   // [B][NH][D][SK]    8 MiB
  __bf16* att = (__bf16*)(ws + 60 * MB);   // [B][SQ][H]        8 MiB
  (void)in_sizes; (void)n_in; (void)out_size; (void)ws_size;

  // stage 0: precision conversion + weight transpose (so WMMA B-frags are contiguous)
  k_cvt_bf16<<<(B_ * SQ_ * H_) / 256, 256, 0, stream>>>(xq, xqb, B_ * SQ_ * H_);
  k_cvt_bf16<<<(B_ * SK_ * H_) / 256, 256, 0, stream>>>(xs, xsb, B_ * SK_ * H_);
  k_transpose_w<<<(3 * H_ * H_) / 256, 256, 0, stream>>>(Wq, wqt, 3 * H_);
  k_transpose_w<<<(3 * H_ * H_) / 256, 256, 0, stream>>>(Wk, wkt, 3 * H_);
  k_transpose_w<<<(3 * H_ * H_) / 256, 256, 0, stream>>>(Wv, wvt, 3 * H_);
  k_transpose_w<<<(H_ * H_) / 256, 256, 0, stream>>>(Wo, wot, H_);

  // stage 1: QKV projections (conv1d 'same' as K=3072 GEMM); Q pre-scaled by D^-0.5
  dim3 gq(SQ_ / 128, H_ / 64, B_);
  k_qkv_gemm<<<gq, 256, 0, stream>>>(xqb, wqt, bq, qb, 0.125f, 0);
  k_qkv_gemm<<<gq, 256, 0, stream>>>(xsb, wkt, bk, kb, 1.0f, 0);
  k_qkv_gemm<<<gq, 256, 0, stream>>>(xsb, wvt, bv, vtb, 1.0f, 1);

  // stage 2: flash attention (TDM K/V tiles, 40 KiB dynamic LDS)
  dim3 ga(SQ_ / 64, NH_, B_);
  k_attn<<<ga, 128, 40960, stream>>>(qb, kb, vtb, mask, att);

  // stage 3: output projection + bias -> f32 output
  dim3 go((B_ * SQ_) / 128, H_ / 64, 1);
  k_out_gemm<<<go, 256, 0, stream>>>(att, wot, bo, out);
}
